// LoRALinear_51608327029057
// MI455X (gfx1250) — matrix operand order repaired and checked by reading, not compiled, not run on hardware
//
#include <hip/hip_runtime.h>

typedef __attribute__((ext_vector_type(16))) __bf16 v16bf;
typedef __attribute__((ext_vector_type(8)))  float  v8f;

#define D_IN   4096
#define D_OUT  4096
#define M_TOT  8192
#define LORA_R 16
#define LORA_SCALING 2.0f   // alpha/rank = 32/16

__device__ __forceinline__ unsigned short f32_to_bf16_rne(float f) {
    unsigned int u = __float_as_uint(f);
    unsigned int r = u + 0x7FFFu + ((u >> 16) & 1u);
    return (unsigned short)(r >> 16);
}
__device__ __forceinline__ float bf16_bits_to_f32(unsigned short h) {
    return __uint_as_float(((unsigned int)h) << 16);
}
__device__ __forceinline__ void split_bf16(float f, unsigned short& h, unsigned short& l) {
    h = f32_to_bf16_rne(f);
    l = f32_to_bf16_rne(f - bf16_bits_to_f32(h));
}

// --------------------------------------------------------------------------
// CDNA5 async global->LDS copy (ASYNCcnt-tracked, no VGPR staging).
// Flat LDS pointer: addr[31:0] is the LDS byte offset (aperture rules, ISA 10.2).
// --------------------------------------------------------------------------
__device__ __forceinline__ unsigned to_lds_off(const void* p) {
    return (unsigned)(unsigned long long)p;
}
__device__ __forceinline__ void async_load_b128(unsigned lds_off, const void* gptr) {
    asm volatile("global_load_async_to_lds_b128 %0, %1, off"
                 :: "v"(lds_off), "v"(gptr)
                 : "memory");
}
__device__ __forceinline__ void wait_asynccnt0() {
    asm volatile("s_wait_asynccnt 0x0" ::: "memory");
}

// --------------------------------------------------------------------------
// Prepass 1: W_eff = W + scaling * (lora_B @ lora_A), split into bf16 hi/lo.
// --------------------------------------------------------------------------
__global__ void __launch_bounds__(256)
lora_fold_split_w(const float* __restrict__ W,
                  const float* __restrict__ lA,   // [R, D_IN]
                  const float* __restrict__ lB,   // [D_OUT, R]
                  unsigned short* __restrict__ Wh,
                  unsigned short* __restrict__ Wl) {
    int idx = blockIdx.x * 256 + threadIdx.x;
    int o   = idx >> 10;
    int d   = (idx & 1023) << 2;

    float bv[LORA_R];
#pragma unroll
    for (int r = 0; r < LORA_R; ++r) bv[r] = lB[(size_t)o * LORA_R + r];

    float4 w = *(const float4*)(W + (size_t)o * D_IN + d);
    float a0 = 0.f, a1 = 0.f, a2 = 0.f, a3 = 0.f;
#pragma unroll
    for (int r = 0; r < LORA_R; ++r) {
        float4 a = *(const float4*)(lA + (size_t)r * D_IN + d);
        a0 += bv[r] * a.x; a1 += bv[r] * a.y;
        a2 += bv[r] * a.z; a3 += bv[r] * a.w;
    }
    w.x += LORA_SCALING * a0; w.y += LORA_SCALING * a1;
    w.z += LORA_SCALING * a2; w.w += LORA_SCALING * a3;

    ushort4 h, l;
    split_bf16(w.x, h.x, l.x); split_bf16(w.y, h.y, l.y);
    split_bf16(w.z, h.z, l.z); split_bf16(w.w, h.w, l.w);
    *(ushort4*)(Wh + (size_t)o * D_IN + d) = h;
    *(ushort4*)(Wl + (size_t)o * D_IN + d) = l;
}

// --------------------------------------------------------------------------
// Prepass 2: split x into bf16 hi/lo (done once, not once per N-block).
// --------------------------------------------------------------------------
__global__ void __launch_bounds__(256)
split_x_bf16(const float* __restrict__ X,
             unsigned short* __restrict__ Xh,
             unsigned short* __restrict__ Xl) {
    size_t d = ((size_t)blockIdx.x * 256 + threadIdx.x) << 2;
    float4 v = *(const float4*)(X + d);
    ushort4 h, l;
    split_bf16(v.x, h.x, l.x); split_bf16(v.y, h.y, l.y);
    split_bf16(v.z, h.z, l.z); split_bf16(v.w, h.w, l.w);
    *(ushort4*)(Xh + d) = h;
    *(ushort4*)(Xl + d) = l;
}

// --------------------------------------------------------------------------
// Fragment loaders from LDS (row-major [row][32] bf16 tiles).
// A (16-bit 16x32): lane half g=lane>>4 holds K = {g*8..g*8+7, 16+g*8..16+g*8+7}
// B (16-bit 32x16): lane half g holds K = {g*8..g*8+7, 16+g*8..16+g*8+7}
// --------------------------------------------------------------------------
__device__ __forceinline__ v16bf load_frag_A(const unsigned short* rowp, int g) {
    union { v16bf v; uint4 q[2]; } u;
    const unsigned short* p = rowp + g * 8;
    u.q[0] = *(const uint4*)(p);
    u.q[1] = *(const uint4*)(p + 16);
    return u.v;
}
__device__ __forceinline__ v16bf load_frag_B(const unsigned short* rowp, int g) {
    union { v16bf v; uint4 q[2]; } u;
    const unsigned short* p = rowp + g * 8;
    u.q[0] = *(const uint4*)(p);
    u.q[1] = *(const uint4*)(p + 16);
    return u.v;
}

#define WMMA_BF16(a, b, c) \
    __builtin_amdgcn_wmma_f32_16x16x32_bf16(false, (a), false, (b), (short)0, (c), false, false)

// Shared compute step: 24 WMMAs from one staged LDS buffer.
// s layout: [Xh | Xl | Wh | Wl], each 128*32 bf16 (4096 ushorts).
__device__ __forceinline__ void compute_step(const unsigned short* s, v8f acc[2][4],
                                             int wm, int wn, int g, int rs) {
    v16bf ah[2], al[2], bh[4], bl[4];
#pragma unroll
    for (int tm = 0; tm < 2; ++tm) {
        int row = wm * 32 + tm * 16 + rs;
        ah[tm] = load_frag_A(s + row * 32, g);
        al[tm] = load_frag_A(s + 4096 + row * 32, g);
    }
#pragma unroll
    for (int tn = 0; tn < 4; ++tn) {
        int row = wn * 64 + tn * 16 + rs;
        bh[tn] = load_frag_B(s + 8192 + row * 32, g);
        bl[tn] = load_frag_B(s + 12288 + row * 32, g);
    }
#pragma unroll
    for (int tm = 0; tm < 2; ++tm)
#pragma unroll
        for (int tn = 0; tn < 4; ++tn) {
            acc[tm][tn] = WMMA_BF16(ah[tm], bh[tn], acc[tm][tn]);
            acc[tm][tn] = WMMA_BF16(ah[tm], bl[tn], acc[tm][tn]);
            acc[tm][tn] = WMMA_BF16(al[tm], bh[tn], acc[tm][tn]);
        }
}

__device__ __forceinline__ void write_out(float* __restrict__ Out,
                                          const float* __restrict__ bias,
                                          v8f acc[2][4], int m_base, int n_base,
                                          int wm, int wn, int g, int rs) {
#pragma unroll
    for (int tm = 0; tm < 2; ++tm)
#pragma unroll
        for (int tn = 0; tn < 4; ++tn) {
            int n = n_base + wn * 64 + tn * 16 + rs;
            float bv = bias[n];
#pragma unroll
            for (int r = 0; r < 8; ++r) {
                int m = m_base + wm * 32 + tm * 16 + g * 8 + r;
                Out[(size_t)m * D_OUT + n] = acc[tm][tn][r] + bv;
            }
        }
}

// --------------------------------------------------------------------------
// Main GEMM (pre-split path): all-bf16 tiles, double-buffered LDS fed by
// ASYNCcnt-tracked global_load_async_to_lds_b128 (no VGPR staging, no spills).
// --------------------------------------------------------------------------
__global__ void __launch_bounds__(256)
lora_gemm_bf16x3_pre(const unsigned short* __restrict__ Xh,
                     const unsigned short* __restrict__ Xl,
                     const unsigned short* __restrict__ Wh,
                     const unsigned short* __restrict__ Wl,
                     const float* __restrict__ bias,
                     float* __restrict__ Out) {
    __shared__ unsigned short smem[2][4 * 128 * 32];   // 2 x 32 KB

    const int tid  = threadIdx.x;
    const int lane = tid & 31;
    const int wave = tid >> 5;
    const int wm   = wave >> 1;     // 0..3
    const int wn   = wave & 1;      // 0..1
    const int g    = lane >> 4;
    const int rs   = lane & 15;

    const int m_base = blockIdx.y * 128;
    const int n_base = blockIdx.x * 128;

    // Per-thread chunks: 2 x 16B per tile (128x32 bf16 = 512 chunks).
    const int row0 = tid >> 2;            const int c0 = (tid & 3) << 3;
    const int row1 = (tid + 256) >> 2;    const int c1 = ((tid + 256) & 3) << 3;

    const unsigned short* srcs[4]  = { Xh, Xl, Wh, Wl };
    const int             rbase[4] = { m_base, m_base, n_base, n_base };

    v8f acc[2][4];
#pragma unroll
    for (int tm = 0; tm < 2; ++tm)
#pragma unroll
        for (int tn = 0; tn < 4; ++tn) acc[tm][tn] = {};

    // --- prologue: async-copy tile 0 into buffer 0 ---
#pragma unroll
    for (int t = 0; t < 4; ++t) {
        async_load_b128(to_lds_off(smem[0] + t * 4096 + row0 * 32 + c0),
                        srcs[t] + (size_t)(rbase[t] + row0) * D_IN + c0);
        async_load_b128(to_lds_off(smem[0] + t * 4096 + row1 * 32 + c1),
                        srcs[t] + (size_t)(rbase[t] + row1) * D_IN + c1);
    }
    wait_asynccnt0();
    __syncthreads();

    const int NK = D_IN / 32;   // 128
    for (int kt = 0; kt < NK; ++kt) {
        if (kt + 1 < NK) {
            // Issue async copy of tile kt+1 into the buffer last read at kt-1
            // (safe: all reads of it drained before the barrier ending kt-1).
            const int k0 = (kt + 1) * 32;
            const int nb = (kt + 1) & 1;
#pragma unroll
            for (int t = 0; t < 4; ++t) {
                async_load_b128(to_lds_off(smem[nb] + t * 4096 + row0 * 32 + c0),
                                srcs[t] + (size_t)(rbase[t] + row0) * D_IN + k0 + c0);
                async_load_b128(to_lds_off(smem[nb] + t * 4096 + row1 * 32 + c1),
                                srcs[t] + (size_t)(rbase[t] + row1) * D_IN + k0 + c1);
            }
        }

        compute_step(smem[kt & 1], acc, wm, wn, g, rs);   // overlaps async copy

        if (kt + 1 < NK) {
            wait_asynccnt0();       // async tile landed in LDS
            __syncthreads();        // publish to all waves
        }
    }

    write_out(Out, bias, acc, m_base, n_base, wm, wn, g, rs);
}

// --------------------------------------------------------------------------
// Fallback GEMM (workspace too small for x split): on-the-fly x conversion.
// --------------------------------------------------------------------------
__global__ void __launch_bounds__(256)
lora_gemm_bf16x3_fly(const float* __restrict__ X,
                     const unsigned short* __restrict__ Wh,
                     const unsigned short* __restrict__ Wl,
                     const float* __restrict__ bias,
                     float* __restrict__ Out) {
    __shared__ unsigned short smem[4 * 128 * 32];   // [Xh|Xl|Wh|Wl]

    const int tid  = threadIdx.x;
    const int lane = tid & 31;
    const int wave = tid >> 5;
    const int wm   = wave >> 1;
    const int wn   = wave & 1;
    const int g    = lane >> 4;
    const int rs   = lane & 15;

    const int m_base = blockIdx.y * 128;
    const int n_base = blockIdx.x * 128;

    v8f acc[2][4];
#pragma unroll
    for (int tm = 0; tm < 2; ++tm)
#pragma unroll
        for (int tn = 0; tn < 4; ++tn) acc[tm][tn] = {};

    for (int k0 = 0; k0 < D_IN; k0 += 32) {
#pragma unroll
        for (int i = 0; i < 4; ++i) {
            int idx = tid + i * 256;
            int row = idx >> 3;
            int c   = (idx & 7) << 2;
            float4 v = *(const float4*)(X + (size_t)(m_base + row) * D_IN + k0 + c);
            ushort4 h, l;
            split_bf16(v.x, h.x, l.x); split_bf16(v.y, h.y, l.y);
            split_bf16(v.z, h.z, l.z); split_bf16(v.w, h.w, l.w);
            *(ushort4*)(smem + row * 32 + c) = h;
            *(ushort4*)(smem + 4096 + row * 32 + c) = l;
        }
#pragma unroll
        for (int i = 0; i < 2; ++i) {
            int idx = tid + i * 256;
            int row = idx >> 2;
            int c   = (idx & 3) << 3;
            const size_t goff = (size_t)(n_base + row) * D_IN + k0 + c;
            *(uint4*)(smem + 8192 + row * 32 + c)  = *(const uint4*)(Wh + goff);
            *(uint4*)(smem + 12288 + row * 32 + c) = *(const uint4*)(Wl + goff);
        }
        __syncthreads();

        compute_step(smem, acc, wm, wn, g, rs);
        __syncthreads();
    }

    write_out(Out, bias, acc, m_base, n_base, wm, wn, g, rs);
}

extern "C" void kernel_launch(void* const* d_in, const int* in_sizes, int n_in,
                              void* d_out, int out_size, void* d_ws, size_t ws_size,
                              hipStream_t stream) {
    const float* x  = (const float*)d_in[0];   // [4, 2048, 4096]
    const float* W  = (const float*)d_in[1];   // [4096, 4096]
    const float* b  = (const float*)d_in[2];   // [4096]
    const float* lA = (const float*)d_in[3];   // [16, 4096]
    const float* lB = (const float*)d_in[4];   // [4096, 16]
    float* out = (float*)d_out;

    const size_t wElems = (size_t)D_OUT * D_IN;   // 16M
    const size_t xElems = (size_t)M_TOT * D_IN;   // 32M
    unsigned short* Wh = (unsigned short*)d_ws;
    unsigned short* Wl = Wh + wElems;

    // Prepass: fold LoRA into W, split to bf16 hi/lo (64 MB of workspace).
    lora_fold_split_w<<<(D_OUT * (D_IN / 4)) / 256, 256, 0, stream>>>(W, lA, lB, Wh, Wl);

    dim3 grid(D_OUT / 128, M_TOT / 128);          // (32, 64)

    const size_t need_full = (2 * wElems + 2 * xElems) * sizeof(unsigned short); // 192 MB
    if (ws_size >= need_full) {
        unsigned short* Xh = Wl + wElems;
        unsigned short* Xl = Xh + xElems;
        split_x_bf16<<<(int)(xElems / 4 / 256), 256, 0, stream>>>(x, Xh, Xl);
        lora_gemm_bf16x3_pre<<<grid, 256, 0, stream>>>(Xh, Xl, Wh, Wl, b, out);
    } else {
        lora_gemm_bf16x3_fly<<<grid, 256, 0, stream>>>(x, Wh, Wl, b, out);
    }
}
